// KAST_72361609003360
// MI455X (gfx1250) — matrix operand order repaired and checked by reading, not compiled, not run on hardware
//
#include <hip/hip_runtime.h>
#include <math.h>

// Flash-attention style kernel for the dual-source softmax-attention reference.
//   B=2, T=8, HW=4096, C=64, Cv=3, M=1024, out = (B, T-1, HW, Cv) fp32.
// One wave32 owns a 32-query tile (two 16-query WMMA column tiles) of one
// (b,t) pair, so each key-tile A fragment (v_perm bf16 pack + loads) and each
// value row is amortized over two 16x16 logit tiles (4 WMMAs per chunk, two
// independent accumulator chains).
// QK^T logits via v_wmma_f32_16x16x32_bf16 (fp32 accumulate); softmax + the
// 3-wide PV in fp32 VALU; exp in exp2 domain (queries pre-scaled by log2 e)
// using the bare v_exp_f32 hardware op. Each half-wave runs an independent
// online softmax over its key subset; states merge once at the end.

#define COEF_MEMORY 0.2f

typedef __attribute__((ext_vector_type(16))) __bf16    v16bf;
typedef __attribute__((ext_vector_type(8)))  float     v8f;
typedef __attribute__((ext_vector_type(8)))  unsigned  v8u;

static __device__ __forceinline__ float fast_exp2(float x) {
#if __has_builtin(__builtin_amdgcn_exp2f)
  return __builtin_amdgcn_exp2f(x);   // bare v_exp_f32
#else
  return exp2f(x);
#endif
}

// Pack two fp32 into two truncated bf16s in one v_perm_b32.
static __device__ __forceinline__ unsigned pack_bf16(float lo, float hi) {
  return __builtin_amdgcn_perm(__builtin_bit_cast(unsigned, hi),
                               __builtin_bit_cast(unsigned, lo),
                               0x07060302u);
}

// Online-softmax attention of this wave's 32 queries (two B-fragment pairs,
// channels 0..31 / 32..63, pre-scaled by log2 e) over `nkeys` keys with
// 3-wide values. Per-half-wave private (m, l, acc); no shuffles here.
static __device__ __forceinline__ void attend(
    const float* __restrict__ keys,   // [nkeys][64]
    const float* __restrict__ vals,   // [nkeys][3]
    int nkeys, const v16bf bq0[2], const v16bf bq1[2], int lane,
    float m[2], float l[2], float acc[2][3])
{
  const int n = lane & 15;   // A-row (key-in-tile); also query col within tile
  const int h = lane >> 4;   // half-wave id
  for (int kt = 0; kt < nkeys; kt += 16) {
    const float* krow = keys + (size_t)(kt + n) * 64;
    if (kt + 16 < nkeys)  // prefetch next key tile (global_prefetch_b8)
      __builtin_prefetch(keys + (size_t)(kt + 16 + n) * 64, 0, 0);

    // A fragments: 16x32 bf16 ISA layout: lane(m=n, half h), VGPR p holds
    // K = (p/4)*16 + h*8 + (p%4)*2 (+1). One v_perm per packed pair.
    v8u ua0, ua1;
#pragma unroll
    for (int p = 0; p < 8; ++p) {
      int kb = ((p >> 2) << 4) + (h << 3) + ((p & 3) << 1);
      ua0[p] = pack_bf16(krow[kb],      krow[kb + 1]);
      ua1[p] = pack_bf16(krow[32 + kb], krow[32 + kb + 1]);
    }
    v16bf a0 = __builtin_bit_cast(v16bf, ua0);
    v16bf a1 = __builtin_bit_cast(v16bf, ua1);

    // Two independent S^T tiles: D(16 keys x 16 queries) per query tile.
    v8f s[2];
#pragma unroll
    for (int tq = 0; tq < 2; ++tq) {
      v8f acc0 = {};
      acc0 = __builtin_amdgcn_wmma_f32_16x16x32_bf16(false, a0, false, bq0[tq],
                                                     (short)0, acc0, false, false);
      s[tq] = __builtin_amdgcn_wmma_f32_16x16x32_bf16(false, a1, false, bq1[tq],
                                                      (short)0, acc0, false, false);
    }
    // s[tq][r] = log2e * logit(key = kt + r + 8h, query = tq*16 + n)

    // Shared value rows for this half's 8 keys (reused by both query tiles).
    const float* vrow = vals + (size_t)(kt + 8 * h) * 3;
    float vv[24];
#pragma unroll
    for (int i = 0; i < 24; ++i) vv[i] = vrow[i];

#pragma unroll
    for (int tq = 0; tq < 2; ++tq) {
      float m01 = fmaxf(s[tq][0], s[tq][1]), m23 = fmaxf(s[tq][2], s[tq][3]);
      float m45 = fmaxf(s[tq][4], s[tq][5]), m67 = fmaxf(s[tq][6], s[tq][7]);
      float cmax = fmaxf(fmaxf(m01, m23), fmaxf(m45, m67));
      float mnew = fmaxf(m[tq], cmax);
      float corr = fast_exp2(m[tq] - mnew);  // exp2(-inf)=0 on 1st tile

      float p8[8];
#pragma unroll
      for (int r = 0; r < 8; ++r) p8[r] = fast_exp2(s[tq][r] - mnew);
      float psum = ((p8[0] + p8[1]) + (p8[2] + p8[3])) +
                   ((p8[4] + p8[5]) + (p8[6] + p8[7]));
      l[tq] = fmaf(l[tq], corr, psum);

#pragma unroll
      for (int d = 0; d < 3; ++d) acc[tq][d] *= corr;
#pragma unroll
      for (int r = 0; r < 8; ++r)
#pragma unroll
        for (int d = 0; d < 3; ++d)
          acc[tq][d] = fmaf(p8[r], vv[r * 3 + d], acc[tq][d]);

      m[tq] = mnew;
    }
  }
}

// Merge the two half-waves' (m, l, acc) softmax states (split-K combine).
static __device__ __forceinline__ void merge_halves(
    float m, float l, const float acc[3], float o[3])
{
  float mo = __shfl_xor(m, 16, 32);
  float mt = fmaxf(m, mo);
  float sc = fast_exp2(m - mt);
  float lt = l * sc;
  lt += __shfl_xor(lt, 16, 32);
#pragma unroll
  for (int d = 0; d < 3; ++d) {
    float a = acc[d] * sc;
    a += __shfl_xor(a, 16, 32);
    o[d] = a / lt;
  }
}

__global__ __launch_bounds__(256) void KAST_72361609003360_kernel(
    const float* __restrict__ k,   // [B][T][HW][C]
    const float* __restrict__ v,   // [B][T][HW][Cv]
    const float* __restrict__ mk,  // [B][T-1][M][C]
    const float* __restrict__ mv,  // [B][T-1][M][Cv]
    float* __restrict__ out)       // [B][T-1][HW][Cv]
{
  constexpr int B = 2, T = 8, HW = 4096, C = 64, Cv = 3, M = 1024;
  constexpr int QT = HW / 32;  // 128 query tiles (32 queries each) per (b,t)
  constexpr float LOG2E = 1.44269504088896340736f;

  const int lane = threadIdx.x & 31;
  const int wid  = blockIdx.x * (blockDim.x >> 5) + (threadIdx.x >> 5);
  const int bt = wid / QT;             // 0..13
  const int qt = wid % QT;
  const int b = bt / (T - 1), t = bt % (T - 1);

  const int n = lane & 15, h = lane >> 4;
  const int q0 = qt * 32;

  const float* kjbase = k + (((size_t)b * T + (t + 1)) * HW + q0 + n) * C;
  const float* ki  = k  + (((size_t)b * T + t) * HW) * C;
  const float* vi  = v  + (((size_t)b * T + t) * HW) * Cv;
  const float* mkp = mk + (((size_t)b * (T - 1) + t) * M) * C;
  const float* mvp = mv + (((size_t)b * (T - 1) + t) * M) * Cv;

  // B fragments for two 16-query tiles, scaled by log2 e, fixed per wave.
  // B layout 32x16 bf16: lane(col n, half h); element j <-> K = (j/8)*16 + h*8 + j%8.
  v16bf bq0[2], bq1[2];
#pragma unroll
  for (int tq = 0; tq < 2; ++tq) {
    const float* kj = kjbase + (size_t)tq * 16 * C;
    v8u u0, u1;
#pragma unroll
    for (int j = 0; j < 8; ++j) {
      int kb = ((j >> 2) << 4) + (h << 3) + ((j & 3) << 1);
      u0[j] = pack_bf16(kj[kb] * LOG2E,
                        kj[kb + 1] * LOG2E);
      u1[j] = pack_bf16(kj[32 + kb] * LOG2E,
                        kj[32 + kb + 1] * LOG2E);
    }
    bq0[tq] = __builtin_bit_cast(v16bf, u0);
    bq1[tq] = __builtin_bit_cast(v16bf, u1);
  }

  float mK[2] = {-INFINITY, -INFINITY}, lK[2] = {0.f, 0.f};
  float accK[2][3] = {{0.f, 0.f, 0.f}, {0.f, 0.f, 0.f}};
  attend(ki, vi, HW, bq0, bq1, lane, mK, lK, accK);

  float mM[2] = {-INFINITY, -INFINITY}, lM[2] = {0.f, 0.f};
  float accM[2][3] = {{0.f, 0.f, 0.f}, {0.f, 0.f, 0.f}};
  attend(mkp, mvp, M, bq0, bq1, lane, mM, lM, accM);

#pragma unroll
  for (int tq = 0; tq < 2; ++tq) {
    float oK[3], oM[3];
    merge_halves(mK[tq], lK[tq], accK[tq], oK);
    merge_halves(mM[tq], lM[tq], accM[tq], oM);
    if (h == 0) {
      float* op = out + ((size_t)bt * HW + q0 + tq * 16 + n) * Cv;
#pragma unroll
      for (int d = 0; d < 3; ++d)
        op[d] = (1.0f - COEF_MEMORY) * oK[d] + COEF_MEMORY * oM[d];
    }
  }
}

extern "C" void kernel_launch(void* const* d_in, const int* in_sizes, int n_in,
                              void* d_out, int out_size, void* d_ws, size_t ws_size,
                              hipStream_t stream) {
  (void)in_sizes; (void)n_in; (void)out_size; (void)d_ws; (void)ws_size;
  const float* k  = (const float*)d_in[0];
  const float* v  = (const float*)d_in[1];
  const float* mk = (const float*)d_in[2];
  const float* mv = (const float*)d_in[3];
  float* out = (float*)d_out;

  constexpr int B = 2, T = 8, HW = 4096;
  constexpr int waves = B * (T - 1) * (HW / 32);  // 1792 wave-tasks
  constexpr int block = 256;                       // 8 wave32 per block
  constexpr int grid  = waves / (block / 32);      // 224 blocks

  KAST_72361609003360_kernel<<<grid, block, 0, stream>>>(k, v, mk, mv, out);
}
